// Decoder_38379827757679
// MI455X (gfx1250) — matrix operand order repaired and checked by reading, not compiled, not run on hardware
//
#include <hip/hip_runtime.h>
#include <hip/hip_bf16.h>
#include <math.h>

typedef __bf16 bf16_t;
typedef __attribute__((ext_vector_type(8)))  __bf16 v8bf;
typedef __attribute__((ext_vector_type(16))) __bf16 v16bf;
typedef __attribute__((ext_vector_type(8)))  float  v8f;

#define LNUM 4
#define HNUM 4
#define BBATCH 2
#define TT 1024
#define DD 1024
#define DFFN 4096
#define VV 32000
#define NTOK (BBATCH * TT)   // 2048

// ---------------------------------------------------------------------------
// Embedding + interleaved sin/cos positional encoding -> h (f32)
// ---------------------------------------------------------------------------
__global__ void embed_kernel(const int* __restrict__ x,
                             const float* __restrict__ tok,
                             float* __restrict__ h) {
  int row = blockIdx.x;            // 0..NTOK-1, row = b*T + t
  int t = row % TT;
  int id = x[row];
  const float* te = tok + (size_t)id * DD;
  float* hr = h + (size_t)row * DD;
  for (int d = threadIdx.x; d < DD; d += blockDim.x) {
    int i = d >> 1;
    float denom = __powf(10000.0f, (2.0f * (float)i) / (float)DD);
    float ang = (float)t / denom;
    float p = (d & 1) ? __cosf(ang) : __sinf(ang);
    hr[d] = te[d] + p;
  }
}

// ---------------------------------------------------------------------------
// LayerNorm (biased var, eps 1e-5) f32 -> bf16
// ---------------------------------------------------------------------------
__global__ void layernorm_kernel(const float* __restrict__ h,
                                 const float* __restrict__ g,
                                 const float* __restrict__ b,
                                 bf16_t* __restrict__ out) {
  int row = blockIdx.x;
  const float* xr = h + (size_t)row * DD;
  __shared__ float r1[256], r2[256];
  float s = 0.f, s2 = 0.f;
  for (int d = threadIdx.x; d < DD; d += 256) {
    float v = xr[d];
    s += v; s2 += v * v;
  }
  r1[threadIdx.x] = s; r2[threadIdx.x] = s2;
  __syncthreads();
  for (int off = 128; off > 0; off >>= 1) {
    if (threadIdx.x < off) {
      r1[threadIdx.x] += r1[threadIdx.x + off];
      r2[threadIdx.x] += r2[threadIdx.x + off];
    }
    __syncthreads();
  }
  float mu  = r1[0] * (1.0f / DD);
  float var = r2[0] * (1.0f / DD) - mu * mu;
  float rstd = rsqrtf(var + 1e-5f);
  bf16_t* o = out + (size_t)row * DD;
  for (int d = threadIdx.x; d < DD; d += 256)
    o[d] = (bf16_t)((xr[d] - mu) * rstd * g[d] + b[d]);
}

// ---------------------------------------------------------------------------
// Tiled transpose + f32->bf16 convert: src[R][C] -> dst[C][R], batched in z
// ---------------------------------------------------------------------------
__global__ void transpose_conv_kernel(const float* __restrict__ src,
                                      bf16_t* __restrict__ dst, int R, int C) {
  __shared__ float tile[32][33];
  size_t bo = (size_t)blockIdx.z * (size_t)R * (size_t)C;
  const float* s = src + bo;
  bf16_t* d = dst + bo;
  int c0 = blockIdx.x * 32, r0 = blockIdx.y * 32;
  for (int i = threadIdx.y; i < 32; i += 8)
    tile[i][threadIdx.x] = s[(size_t)(r0 + i) * C + (c0 + threadIdx.x)];
  __syncthreads();
  for (int i = threadIdx.y; i < 32; i += 8)
    d[(size_t)(c0 + i) * R + (r0 + threadIdx.x)] = (bf16_t)tile[threadIdx.x][i];
}

// ---------------------------------------------------------------------------
// Generic WMMA bf16 GEMM:  C[M,N] = A[M,K] * BT[N,K]^T  (+bias, +epilogue)
// Block tile 128x128, 8 waves in a 4x2 grid; each wave owns a 32x64 subtile
// (2 A-frags x 4 B-frags -> 8 accumulators). LDS double-buffered, register-
// staged global prefetch; one barrier per K-step; fragments preloaded so the
// 8 WMMAs issue back-to-back.
// ---------------------------------------------------------------------------
#define BM 128
#define BN 128
#define BK 32

#define MAKE_FRAG(lo, hi) __builtin_shufflevector(lo, hi, 0, 1, 2, 3, 4, 5, 6, 7, \
                                                  8, 9, 10, 11, 12, 13, 14, 15)

__global__ __launch_bounds__(256)
void wmma_gemm_kernel(const bf16_t* __restrict__ A, int lda,
                      const bf16_t* __restrict__ BT, int ldb,
                      float* outF, bf16_t* outB, int ldc,
                      int K,
                      const float* bias,
                      int act, int resid, int causal, int tstore,
                      long strideA, long strideB, long strideC, long strideBias) {
  int zb = blockIdx.z;
  A  += (size_t)zb * (size_t)strideA;
  BT += (size_t)zb * (size_t)strideB;
  if (outF) outF += (size_t)zb * (size_t)strideC;
  if (outB) outB += (size_t)zb * (size_t)strideC;
  if (bias) bias += (size_t)zb * (size_t)strideBias;

  int n0 = blockIdx.x * BN;
  int m0 = blockIdx.y * BM;
  if (causal && (n0 >= m0 + BM)) return;   // tile entirely above causal diagonal

  __shared__ alignas(16) bf16_t Asub[2][BM][40];   // 80B row pitch, 16B aligned
  __shared__ alignas(16) bf16_t Bsub[2][BN][40];

  int tid  = threadIdx.x;
  int lane = tid & 31;
  int wave = tid >> 5;
  int lr = lane & 15;
  int lh = lane >> 4;
  int mrow = (wave >> 1) * 32;     // 0,32,64,96
  int ncol = (wave & 1) * 64;      // 0,64

  // fixed per-thread global staging coordinates (two 8-elt chunks per matrix)
  int ldrow = tid >> 2;            // 0..63
  int ldcol = (tid & 3) << 3;      // 0,8,16,24
  const bf16_t* gA0 = &A[(size_t)(m0 + ldrow) * lda + ldcol];
  const bf16_t* gA1 = &A[(size_t)(m0 + ldrow + 64) * lda + ldcol];
  const bf16_t* gB0 = &BT[(size_t)(n0 + ldrow) * ldb + ldcol];
  const bf16_t* gB1 = &BT[(size_t)(n0 + ldrow + 64) * ldb + ldcol];

  v8f zero = {0.f, 0.f, 0.f, 0.f, 0.f, 0.f, 0.f, 0.f};
  v8f acc[8];
#pragma unroll
  for (int i = 0; i < 8; i++) acc[i] = zero;

  // preload tile 0 into buffer 0
  v8bf va0 = *(const v8bf*)gA0;
  v8bf va1 = *(const v8bf*)gA1;
  v8bf vb0 = *(const v8bf*)gB0;
  v8bf vb1 = *(const v8bf*)gB1;
  *(v8bf*)&Asub[0][ldrow][ldcol]      = va0;
  *(v8bf*)&Asub[0][ldrow + 64][ldcol] = va1;
  *(v8bf*)&Bsub[0][ldrow][ldcol]      = vb0;
  *(v8bf*)&Bsub[0][ldrow + 64][ldcol] = vb1;
  __syncthreads();

  int KT = K / BK;
  for (int kt = 0; kt < KT; kt++) {
    int cur = kt & 1;

    // prefetch next K-tile from global into registers (overlaps with WMMAs)
    bool more = (kt + 1 < KT);
    if (more) {
      int ko = (kt + 1) * BK;
      va0 = *(const v8bf*)(gA0 + ko);
      va1 = *(const v8bf*)(gA1 + ko);
      vb0 = *(const v8bf*)(gB0 + ko);
      vb1 = *(const v8bf*)(gB1 + ko);
    }

    // load all fragments for this wave's 32x64 subtile
    int am0 = mrow + lr;
    int am1 = mrow + 16 + lr;
    v16bf afr0 = MAKE_FRAG(*(const v8bf*)&Asub[cur][am0][8 * lh],
                           *(const v8bf*)&Asub[cur][am0][16 + 8 * lh]);
    v16bf afr1 = MAKE_FRAG(*(const v8bf*)&Asub[cur][am1][8 * lh],
                           *(const v8bf*)&Asub[cur][am1][16 + 8 * lh]);
    v16bf bfr[4];
#pragma unroll
    for (int nt = 0; nt < 4; nt++) {
      int bn = ncol + nt * 16 + lr;
      bfr[nt] = MAKE_FRAG(*(const v8bf*)&Bsub[cur][bn][8 * lh],
                          *(const v8bf*)&Bsub[cur][bn][16 + 8 * lh]);
    }

    // 8 back-to-back WMMAs
#pragma unroll
    for (int nt = 0; nt < 4; nt++) {
      acc[nt]     = __builtin_amdgcn_wmma_f32_16x16x32_bf16(
          false, afr0, false, bfr[nt], (short)0, acc[nt], false, false);
      acc[4 + nt] = __builtin_amdgcn_wmma_f32_16x16x32_bf16(
          false, afr1, false, bfr[nt], (short)0, acc[4 + nt], false, false);
    }

    if (more) {
      int nxt = (kt + 1) & 1;
      *(v8bf*)&Asub[nxt][ldrow][ldcol]      = va0;
      *(v8bf*)&Asub[nxt][ldrow + 64][ldcol] = va1;
      *(v8bf*)&Bsub[nxt][ldrow][ldcol]      = vb0;
      *(v8bf*)&Bsub[nxt][ldrow + 64][ldcol] = vb1;
      __syncthreads();
    }
  }

  // Epilogue. C layout per 16x16 tile: lane N = lr, VGPR r -> M = r + 8*lh.
#pragma unroll
  for (int a = 0; a < 2; a++) {
#pragma unroll
    for (int nt = 0; nt < 4; nt++) {
      int n = n0 + ncol + nt * 16 + lr;
      float bv = bias ? bias[n] : 0.f;
#pragma unroll
      for (int r = 0; r < 8; r++) {
        int m = m0 + mrow + a * 16 + r + 8 * lh;
        float v = acc[a * 4 + nt][r] + bv;
        if (act == 1) v = 0.5f * v * (1.0f + erff(v * 0.70710678118f));  // exact GELU
        if (outF) {
          if (resid) outF[(size_t)m * ldc + n] += v;
          else       outF[(size_t)m * ldc + n] = v;
        } else {
          if (tstore) outB[(size_t)n * ldc + m] = (bf16_t)v;  // transposed store (V^T)
          else        outB[(size_t)m * ldc + n] = (bf16_t)v;
        }
      }
    }
  }
}

// ---------------------------------------------------------------------------
// Causal softmax over scaled scores: S[bh][i][j] f32 -> P bf16 (0 for j>i)
// ---------------------------------------------------------------------------
__global__ void softmax_kernel(const float* __restrict__ S,
                               bf16_t* __restrict__ P, float scale) {
  int row = blockIdx.x;                 // query index i
  int bh = blockIdx.y;                  // b*H + h
  const float* s = S + ((size_t)bh * TT + row) * TT;
  bf16_t* p = P + ((size_t)bh * TT + row) * TT;
  int valid = row + 1;
  __shared__ float red[256];

  float mx = -3.4e38f;
  for (int j = threadIdx.x; j < valid; j += 256) mx = fmaxf(mx, s[j] * scale);
  red[threadIdx.x] = mx;
  __syncthreads();
  for (int off = 128; off > 0; off >>= 1) {
    if (threadIdx.x < off)
      red[threadIdx.x] = fmaxf(red[threadIdx.x], red[threadIdx.x + off]);
    __syncthreads();
  }
  mx = red[0];
  __syncthreads();

  float sum = 0.f;
  for (int j = threadIdx.x; j < valid; j += 256) sum += __expf(s[j] * scale - mx);
  red[threadIdx.x] = sum;
  __syncthreads();
  for (int off = 128; off > 0; off >>= 1) {
    if (threadIdx.x < off) red[threadIdx.x] += red[threadIdx.x + off];
    __syncthreads();
  }
  float inv = 1.0f / red[0];

  for (int j = threadIdx.x; j < TT; j += 256)
    p[j] = (j < valid) ? (bf16_t)(__expf(s[j] * scale - mx) * inv) : (bf16_t)0.0f;
}

// ---------------------------------------------------------------------------
// In-place log-softmax over V=32000 per row
// ---------------------------------------------------------------------------
__global__ void logsoftmax_kernel(float* __restrict__ out) {
  int row = blockIdx.x;
  float* x = out + (size_t)row * VV;
  __shared__ float red[256];

  float mx = -3.4e38f;
  for (int j = threadIdx.x; j < VV; j += 256) mx = fmaxf(mx, x[j]);
  red[threadIdx.x] = mx;
  __syncthreads();
  for (int off = 128; off > 0; off >>= 1) {
    if (threadIdx.x < off)
      red[threadIdx.x] = fmaxf(red[threadIdx.x], red[threadIdx.x + off]);
    __syncthreads();
  }
  mx = red[0];
  __syncthreads();

  float sum = 0.f;
  for (int j = threadIdx.x; j < VV; j += 256) sum += __expf(x[j] - mx);
  red[threadIdx.x] = sum;
  __syncthreads();
  for (int off = 128; off > 0; off >>= 1) {
    if (threadIdx.x < off) red[threadIdx.x] += red[threadIdx.x + off];
    __syncthreads();
  }
  float lse = mx + __logf(red[0]);

  for (int j = threadIdx.x; j < VV; j += 256) x[j] -= lse;
}

// ---------------------------------------------------------------------------
// Host orchestration
// ---------------------------------------------------------------------------
extern "C" void kernel_launch(void* const* d_in, const int* in_sizes, int n_in,
                              void* d_out, int out_size, void* d_ws, size_t ws_size,
                              hipStream_t stream) {
  (void)in_sizes; (void)n_in; (void)out_size; (void)ws_size;

  const int*   x_ids    = (const int*)  d_in[0];
  const float* token_emb= (const float*)d_in[1];
  const float* Wq  = (const float*)d_in[2];
  const float* bq  = (const float*)d_in[3];
  const float* Wk  = (const float*)d_in[4];
  const float* bk  = (const float*)d_in[5];
  const float* Wv  = (const float*)d_in[6];
  const float* bv  = (const float*)d_in[7];
  const float* Wo  = (const float*)d_in[8];
  const float* bo  = (const float*)d_in[9];
  const float* ln1g= (const float*)d_in[10];
  const float* ln1b= (const float*)d_in[11];
  const float* ln2g= (const float*)d_in[12];
  const float* ln2b= (const float*)d_in[13];
  const float* W1  = (const float*)d_in[14];
  const float* b1  = (const float*)d_in[15];
  const float* W2  = (const float*)d_in[16];
  const float* b2  = (const float*)d_in[17];
  const float* flng= (const float*)d_in[18];
  const float* flnb= (const float*)d_in[19];
  const float* Wout= (const float*)d_in[20];
  float* out = (float*)d_out;

  // ---- workspace carve-up ----
  char* wsp = (char*)d_ws;
  size_t off = 0;
  auto alloc = [&](size_t bytes) -> void* {
    void* p = wsp + off;
    off = (off + bytes + 255) & ~(size_t)255;
    return p;
  };
  const size_t DxD = (size_t)DD * DD;
  bf16_t* wqT   = (bf16_t*)alloc((size_t)LNUM * HNUM * DxD * 2);
  bf16_t* wkT   = (bf16_t*)alloc((size_t)LNUM * HNUM * DxD * 2);
  bf16_t* wvT   = (bf16_t*)alloc((size_t)LNUM * HNUM * DxD * 2);
  bf16_t* woT   = (bf16_t*)alloc((size_t)LNUM * DD * (HNUM * DD) * 2);
  bf16_t* w1T   = (bf16_t*)alloc((size_t)LNUM * DFFN * DD * 2);
  bf16_t* w2T   = (bf16_t*)alloc((size_t)LNUM * DD * DFFN * 2);
  bf16_t* woutT = (bf16_t*)alloc((size_t)VV * DD * 2);
  float*  hbuf  = (float*) alloc((size_t)NTOK * DD * 4);
  bf16_t* xn    = (bf16_t*)alloc((size_t)NTOK * DD * 2);
  bf16_t* Qbuf  = (bf16_t*)alloc((size_t)HNUM * NTOK * DD * 2);
  bf16_t* Kbuf  = (bf16_t*)alloc((size_t)HNUM * NTOK * DD * 2);
  bf16_t* Vt    = (bf16_t*)alloc((size_t)HNUM * DD * NTOK * 2);   // [h][d][b*T+t]
  float*  Sbuf  = (float*) alloc((size_t)BBATCH * HNUM * TT * TT * 4);
  bf16_t* Ap    = (bf16_t*)alloc((size_t)BBATCH * HNUM * TT * TT * 2);
  bf16_t* ybuf  = (bf16_t*)alloc((size_t)NTOK * (HNUM * DD) * 2);
  bf16_t* mbuf  = (bf16_t*)alloc((size_t)NTOK * DFFN * 2);

  auto gemm = [&](const bf16_t* A, int lda, const bf16_t* BT, int ldb,
                  float* oF, bf16_t* oB, int ldc, int M, int N, int K,
                  const float* bias, int act, int resid, int causal, int tstore,
                  int batch, long sA, long sB, long sC, long sBias) {
    dim3 grid(N / BN, M / BM, batch);
    wmma_gemm_kernel<<<grid, dim3(256), 0, stream>>>(
        A, lda, BT, ldb, oF, oB, ldc, K, bias, act, resid, causal, tstore,
        sA, sB, sC, sBias);
  };

  // ---- convert + transpose all weights to bf16 (B^T layout: [N][K]) ----
  {
    dim3 blk(32, 8);
    transpose_conv_kernel<<<dim3(DD/32, DD/32, LNUM*HNUM), blk, 0, stream>>>(Wq, wqT, DD, DD);
    transpose_conv_kernel<<<dim3(DD/32, DD/32, LNUM*HNUM), blk, 0, stream>>>(Wk, wkT, DD, DD);
    transpose_conv_kernel<<<dim3(DD/32, DD/32, LNUM*HNUM), blk, 0, stream>>>(Wv, wvT, DD, DD);
    transpose_conv_kernel<<<dim3(DD/32, (HNUM*DD)/32, LNUM), blk, 0, stream>>>(Wo, woT, HNUM*DD, DD);
    transpose_conv_kernel<<<dim3(DFFN/32, DD/32, LNUM), blk, 0, stream>>>(W1, w1T, DD, DFFN);
    transpose_conv_kernel<<<dim3(DD/32, DFFN/32, LNUM), blk, 0, stream>>>(W2, w2T, DFFN, DD);
    transpose_conv_kernel<<<dim3(VV/32, DD/32, 1), blk, 0, stream>>>(Wout, woutT, DD, VV);
  }

  // ---- embedding ----
  embed_kernel<<<dim3(NTOK), dim3(256), 0, stream>>>(x_ids, token_emb, hbuf);

  const float scale = 0.03125f;  // 1/sqrt(1024)

  for (int l = 0; l < LNUM; l++) {
    // LN1
    layernorm_kernel<<<dim3(NTOK), dim3(256), 0, stream>>>(
        hbuf, ln1g + (size_t)l * DD, ln1b + (size_t)l * DD, xn);

    // Q, K, V projections, batched over heads in grid.z
    const bf16_t* wq_l = wqT + (size_t)l * HNUM * DxD;
    const bf16_t* wk_l = wkT + (size_t)l * HNUM * DxD;
    const bf16_t* wv_l = wvT + (size_t)l * HNUM * DxD;
    const float*  bq_l = bq + (size_t)l * HNUM * DD;
    const float*  bk_l = bk + (size_t)l * HNUM * DD;
    const float*  bv_l = bv + (size_t)l * HNUM * DD;
    gemm(xn, DD, wq_l, DD, nullptr, Qbuf, DD, NTOK, DD, DD,
         bq_l, 0, 0, 0, 0, HNUM, 0, (long)DxD, (long)NTOK * DD, DD);
    gemm(xn, DD, wk_l, DD, nullptr, Kbuf, DD, NTOK, DD, DD,
         bk_l, 0, 0, 0, 0, HNUM, 0, (long)DxD, (long)NTOK * DD, DD);
    // V stored transposed: Vt[h][d][b*T+t], ldc = NTOK
    gemm(xn, DD, wv_l, DD, nullptr, Vt, NTOK, NTOK, DD, DD,
         bv_l, 0, 0, 0, 1, HNUM, 0, (long)DxD, (long)DD * NTOK, DD);

    // S = Q K^T (causal), per batch b, heads batched in z
    for (int b = 0; b < BBATCH; b++) {
      gemm(Qbuf + (size_t)b * TT * DD, DD,
           Kbuf + (size_t)b * TT * DD, DD,
           Sbuf + (size_t)b * HNUM * TT * TT, nullptr, TT,
           TT, TT, DD, nullptr, 0, 0, /*causal=*/1, 0,
           HNUM, (long)NTOK * DD, (long)NTOK * DD, (long)TT * TT, 0);
    }

    // softmax -> bf16 probs
    softmax_kernel<<<dim3(TT, BBATCH * HNUM), dim3(256), 0, stream>>>(Sbuf, Ap, scale);

    // Y = A V  ->  y[b][t][h*D + d]
    for (int b = 0; b < BBATCH; b++) {
      gemm(Ap + (size_t)b * HNUM * TT * TT, TT,
           Vt + (size_t)b * TT, NTOK,
           nullptr, ybuf + (size_t)b * TT * (HNUM * DD), HNUM * DD,
           TT, DD, TT, nullptr, 0, 0, 0, 0,
           HNUM, (long)TT * TT, (long)DD * NTOK, (long)DD, 0);
    }

    // h += y @ Wo + bo
    gemm(ybuf, HNUM * DD, woT + (size_t)l * DD * (HNUM * DD), HNUM * DD,
         hbuf, nullptr, DD, NTOK, DD, HNUM * DD,
         bo + (size_t)l * DD, 0, /*resid=*/1, 0, 0, 1, 0, 0, 0, 0);

    // LN2
    layernorm_kernel<<<dim3(NTOK), dim3(256), 0, stream>>>(
        hbuf, ln2g + (size_t)l * DD, ln2b + (size_t)l * DD, xn);

    // m = gelu(xn @ W1 + b1)
    gemm(xn, DD, w1T + (size_t)l * DFFN * DD, DD,
         nullptr, mbuf, DFFN, NTOK, DFFN, DD,
         b1 + (size_t)l * DFFN, /*act=*/1, 0, 0, 0, 1, 0, 0, 0, 0);

    // h += m @ W2 + b2
    gemm(mbuf, DFFN, w2T + (size_t)l * DD * DFFN, DFFN,
         hbuf, nullptr, DD, NTOK, DD, DFFN,
         b2 + (size_t)l * DD, 0, /*resid=*/1, 0, 0, 1, 0, 0, 0, 0);
  }

  // final LN
  layernorm_kernel<<<dim3(NTOK), dim3(256), 0, stream>>>(hbuf, flng, flnb, xn);

  // logits = xn @ Wout  (f32, straight into d_out)
  gemm(xn, DD, woutT, DD, out, nullptr, VV, NTOK, VV, DD,
       nullptr, 0, 0, 0, 0, 1, 0, 0, 0, 0);

  // log_softmax in place
  logsoftmax_kernel<<<dim3(NTOK), dim3(256), 0, stream>>>(out);
}
